// EncoderLayer_69741678952757
// MI455X (gfx1250) — matrix operand order repaired and checked by reading, not compiled, not run on hardware
//
#include <hip/hip_runtime.h>

#define B_        8
#define T_        1024
#define CONV_DIM_ 256
#define D_MODEL_  1024
#define H_        16
#define DH_       64
#define D_FF_     5120
#define EMBED_    1280
#define EPS_      1e-6f

typedef __attribute__((ext_vector_type(16))) __bf16       v16bf;
typedef __attribute__((ext_vector_type(8)))  float        v8f;
typedef __attribute__((ext_vector_type(4)))  unsigned int u32x4;

union Frag16  { u32x4 q[2]; v16bf v; };
union PackBf2 { unsigned int u; __bf16 h[2]; };

__device__ inline v8f zero_v8f() {
  v8f z;
#pragma unroll
  for (int i = 0; i < 8; ++i) z[i] = 0.f;
  return z;
}

__device__ inline v8f wmma_bf16(v16bf a, v16bf b, v8f c) {
  return __builtin_amdgcn_wmma_f32_16x16x32_bf16(false, a, false, b, (short)0, c,
                                                 false, false);
}

// ---------------------------------------------------------------------------
// Weight convert + transpose: out[n*K + k] = (bf16) in[k*N + n]
// ---------------------------------------------------------------------------
__global__ __launch_bounds__(256) void transpose_convert_kernel(
    const float* __restrict__ in, __bf16* __restrict__ out, int K, int N) {
  __shared__ float tile[32][33];
  const int n0 = blockIdx.x * 32, k0 = blockIdx.y * 32;
  const int tx = threadIdx.x & 31, ty = threadIdx.x >> 5;  // ty 0..7
#pragma unroll
  for (int i = 0; i < 4; ++i)
    tile[ty + i * 8][tx] = in[(size_t)(k0 + ty + i * 8) * N + n0 + tx];
  __syncthreads();
#pragma unroll
  for (int i = 0; i < 4; ++i)
    out[(size_t)(n0 + ty + i * 8) * K + k0 + tx] = (__bf16)tile[tx][ty + i * 8];
}

// att_in (columns 256..1279 of x) -> bf16, row-major (B*T, 1024)
__global__ __launch_bounds__(256) void convert_attin_kernel(
    const float* __restrict__ x, __bf16* __restrict__ o) {
  const size_t i = (size_t)blockIdx.x * 256 + threadIdx.x;  // over B*T*1024
  const size_t m = i >> 10;
  const int    c = (int)(i & 1023);
  o[i] = (__bf16)x[m * EMBED_ + CONV_DIM_ + c];
}

// depthwise conv, K=3, same kernel across all channels
__global__ __launch_bounds__(256) void conv_kernel(
    const float* __restrict__ x, const float* __restrict__ w,
    float* __restrict__ out) {
  const size_t i  = (size_t)blockIdx.x * 256 + threadIdx.x;  // over B*T*256
  const int    c  = (int)(i & (CONV_DIM_ - 1));
  const size_t bt = i >> 8;
  const int    t  = (int)(bt & (T_ - 1));
  const float* xr = x + bt * EMBED_ + c;
  float s = w[1] * xr[0];
  if (t > 0)      s += w[0] * xr[-EMBED_];
  if (t < T_ - 1) s += w[2] * xr[EMBED_];
  out[i] = s;
}

// ---------------------------------------------------------------------------
// Generic bf16 WMMA GEMM: C(MxN) = A(MxK) * Bt(NxK)^T + bias
// mode 0: bf16 out, reshaped (B*T, H*DH) -> (B,H,T,DH)     (QKV projections)
// mode 1: f32 out row-major                                 (attn-out, FFN2)
// mode 2: relu -> bf16 out row-major                        (FFN1)
// Block: 128x128 tile, KT=32, 256 threads (8 wave32, 4x2), wave tile 32x64.
// ---------------------------------------------------------------------------
__global__ __launch_bounds__(256) void gemm_bf16_kernel(
    const __bf16* __restrict__ A, const __bf16* __restrict__ Bt,
    const float* __restrict__ bias, void* __restrict__ out,
    int M, int N, int Kd, int mode) {
  __shared__ __attribute__((aligned(16))) __bf16 As[128][32];
  __shared__ __attribute__((aligned(16))) __bf16 Bs[128][32];

  const int tid  = threadIdx.x;
  const int lane = tid & 31;
  const int wave = tid >> 5;
  const int wm0  = (wave >> 1) * 32;  // wave M origin in block tile
  const int wn0  = (wave & 1) * 64;   // wave N origin in block tile
  const int m0   = blockIdx.y * 128;
  const int n0   = blockIdx.x * 128;
  const int hl   = lane & 15;
  const int kA   = (lane >> 4) << 3;   // A frag K base: 0 / 8
  const int kB   = (lane >> 4) << 3;   // B frag K base: 0 / 8
  const int mr   = (lane >> 4) << 3;   // C frag row base

  v8f acc[2][4];
#pragma unroll
  for (int mi = 0; mi < 2; ++mi)
#pragma unroll
    for (int ni = 0; ni < 4; ++ni) acc[mi][ni] = zero_v8f();

  for (int k0 = 0; k0 < Kd; k0 += 32) {
#pragma unroll
    for (int i = 0; i < 2; ++i) {
      const int idx = tid * 2 + i;      // 0..511 over 128x32 tile (8 bf16 units)
      const int r = idx >> 2;
      const int c = (idx & 3) << 3;
      *(u32x4*)(&As[r][c]) = *(const u32x4*)(A + (size_t)(m0 + r) * Kd + k0 + c);
      *(u32x4*)(&Bs[r][c]) = *(const u32x4*)(Bt + (size_t)(n0 + r) * Kd + k0 + c);
    }
    if (k0 + 32 < Kd) {
      __builtin_prefetch(A + (size_t)(m0 + (tid >> 1)) * Kd + k0 + 32, 0, 1);
      __builtin_prefetch(Bt + (size_t)(n0 + (tid >> 1)) * Kd + k0 + 32, 0, 1);
    }
    __syncthreads();

    Frag16 a[2], b[4];
#pragma unroll
    for (int mi = 0; mi < 2; ++mi) {
      a[mi].q[0] = *(const u32x4*)(&As[wm0 + mi * 16 + hl][kA]);
      a[mi].q[1] = *(const u32x4*)(&As[wm0 + mi * 16 + hl][kA + 16]);
    }
#pragma unroll
    for (int ni = 0; ni < 4; ++ni) {
      b[ni].q[0] = *(const u32x4*)(&Bs[wn0 + ni * 16 + hl][kB]);
      b[ni].q[1] = *(const u32x4*)(&Bs[wn0 + ni * 16 + hl][kB + 16]);
    }
#pragma unroll
    for (int mi = 0; mi < 2; ++mi)
#pragma unroll
      for (int ni = 0; ni < 4; ++ni)
        acc[mi][ni] = wmma_bf16(a[mi].v, b[ni].v, acc[mi][ni]);
    __syncthreads();
  }

#pragma unroll
  for (int mi = 0; mi < 2; ++mi) {
#pragma unroll
    for (int ni = 0; ni < 4; ++ni) {
      const int   n  = n0 + wn0 + ni * 16 + hl;
      const float bv = bias[n];
#pragma unroll
      for (int r = 0; r < 8; ++r) {
        const int   m   = m0 + wm0 + mi * 16 + mr + r;
        const float val = acc[mi][ni][r] + bv;
        if (mode == 0) {  // -> (B,H,T,DH) bf16
          const int bb = m >> 10, t = m & (T_ - 1);
          const int hh = n >> 6, d = n & (DH_ - 1);
          ((__bf16*)out)[(((size_t)bb * H_ + hh) * T_ + t) * DH_ + d] = (__bf16)val;
        } else if (mode == 1) {
          ((float*)out)[(size_t)m * N + n] = val;
        } else {
          ((__bf16*)out)[(size_t)m * N + n] = (__bf16)fmaxf(val, 0.f);
        }
      }
    }
  }
}

// ---------------------------------------------------------------------------
// Flash attention: block = (64 q-rows, head, batch); 4 wave32, 16 rows/wave.
// q,k,v are bf16 (B,H,T,DH); output att bf16 (B,T,H*DH).
// ---------------------------------------------------------------------------
__global__ __launch_bounds__(128) void attention_kernel(
    const __bf16* __restrict__ q, const __bf16* __restrict__ k,
    const __bf16* __restrict__ v, const int* __restrict__ att_mask,
    __bf16* __restrict__ att_out) {
  __shared__ __attribute__((aligned(16))) __bf16 Ks[64][64];      // [key][d]
  __shared__ __attribute__((aligned(16))) __bf16 Vt[64][64];      // [d][key]
  __shared__ __attribute__((aligned(16))) __bf16 Ps[4][16][64];   // per-wave P

  const int tid = threadIdx.x, lane = tid & 31, wave = tid >> 5;
  const int qt = blockIdx.x, h = blockIdx.y, b = blockIdx.z;
  const size_t bh = (size_t)b * H_ + h;
  const __bf16* qg = q + bh * (size_t)(T_ * DH_);
  const __bf16* kg = k + bh * (size_t)(T_ * DH_);
  const __bf16* vg = v + bh * (size_t)(T_ * DH_);
  const int hl = lane & 15;
  const int kA = (lane >> 4) << 3;
  const int kB = (lane >> 4) << 3;
  const int mr = (lane >> 4) << 3;

  // Q fragments for this wave's 16 rows (held in registers for whole kernel)
  const int qrow = qt * 64 + wave * 16 + hl;
  Frag16 qa[2];
#pragma unroll
  for (int ks = 0; ks < 2; ++ks) {
    qa[ks].q[0] = *(const u32x4*)(qg + (size_t)qrow * DH_ + ks * 32 + kA);
    qa[ks].q[1] = *(const u32x4*)(qg + (size_t)qrow * DH_ + ks * 32 + kA + 16);
  }

  float mrow[8], lsum[8];
  v8f   o[4];
#pragma unroll
  for (int r = 0; r < 8; ++r) { mrow[r] = -3.0e38f; lsum[r] = 0.f; }
#pragma unroll
  for (int nf = 0; nf < 4; ++nf) o[nf] = zero_v8f();

  const float scale = 0.125f;  // 1/sqrt(DH)

  for (int j = 0; j < T_ / 64; ++j) {
    // stage K tile [key][d]
#pragma unroll
    for (int i = 0; i < 4; ++i) {
      const int idx = tid * 4 + i;
      const int r = idx >> 3, c = (idx & 7) << 3;
      *(u32x4*)(&Ks[r][c]) = *(const u32x4*)(kg + (size_t)(j * 64 + r) * DH_ + c);
    }
    // stage V transposed [d][key]
#pragma unroll
    for (int i = 0; i < 16; ++i) {
      const int e  = tid + i * 128;   // dword id, 0..2047
      const int tk = e >> 5;
      const int dp = (e & 31) << 1;
      PackBf2 p;
      p.u = *(const unsigned int*)(vg + (size_t)(j * 64 + tk) * DH_ + dp);
      Vt[dp][tk]     = p.h[0];
      Vt[dp + 1][tk] = p.h[1];
    }
    __syncthreads();

    int msk[4];
#pragma unroll
    for (int nf = 0; nf < 4; ++nf)
      msk[nf] = att_mask[b * T_ + j * 64 + nf * 16 + hl];

    // S = Q * K^T (16 x 64), scale, mask (reference: where mask!=0 -> -1e9)
    v8f s[4];
#pragma unroll
    for (int nf = 0; nf < 4; ++nf) {
      s[nf] = zero_v8f();
#pragma unroll
      for (int ks = 0; ks < 2; ++ks) {
        Frag16 kf;
        kf.q[0] = *(const u32x4*)(&Ks[nf * 16 + hl][ks * 32 + kB]);
        kf.q[1] = *(const u32x4*)(&Ks[nf * 16 + hl][ks * 32 + kB + 16]);
        s[nf] = wmma_bf16(qa[ks].v, kf.v, s[nf]);
      }
#pragma unroll
      for (int r = 0; r < 8; ++r)
        s[nf][r] = msk[nf] ? -1.0e9f : s[nf][r] * scale;
    }

    // online softmax: row max / sum via half-wave shuffles (lanes share rows)
    float corr[8], rs[8];
#pragma unroll
    for (int r = 0; r < 8; ++r) {
      float t = fmaxf(fmaxf(s[0][r], s[1][r]), fmaxf(s[2][r], s[3][r]));
#pragma unroll
      for (int d = 1; d < 16; d <<= 1) t = fmaxf(t, __shfl_xor(t, d, 32));
      const float mn = fmaxf(mrow[r], t);
      corr[r] = __expf(mrow[r] - mn);
      mrow[r] = mn;
      rs[r]   = 0.f;
    }
#pragma unroll
    for (int nf = 0; nf < 4; ++nf)
#pragma unroll
      for (int r = 0; r < 8; ++r) {
        const float p = __expf(s[nf][r] - mrow[r]);
        rs[r] += p;
        Ps[wave][mr + r][nf * 16 + hl] = (__bf16)p;  // transpose via LDS
      }
#pragma unroll
    for (int r = 0; r < 8; ++r) {
      float t = rs[r];
#pragma unroll
      for (int d = 1; d < 16; d <<= 1) t += __shfl_xor(t, d, 32);
      lsum[r] = lsum[r] * corr[r] + t;
    }
#pragma unroll
    for (int nf = 0; nf < 4; ++nf)
#pragma unroll
      for (int r = 0; r < 8; ++r) o[nf][r] *= corr[r];

    // wave-local RAW on Ps (LDS): wait for our ds stores before frag loads
    asm volatile("s_wait_dscnt 0" ::: "memory");

    // O += P * V
#pragma unroll
    for (int ks = 0; ks < 2; ++ks) {
      Frag16 pa;
      pa.q[0] = *(const u32x4*)(&Ps[wave][hl][ks * 32 + kA]);
      pa.q[1] = *(const u32x4*)(&Ps[wave][hl][ks * 32 + kA + 16]);
#pragma unroll
      for (int nf = 0; nf < 4; ++nf) {
        Frag16 vf;
        vf.q[0] = *(const u32x4*)(&Vt[nf * 16 + hl][ks * 32 + kB]);
        vf.q[1] = *(const u32x4*)(&Vt[nf * 16 + hl][ks * 32 + kB + 16]);
        o[nf] = wmma_bf16(pa.v, vf.v, o[nf]);
      }
    }
    __syncthreads();
  }

  // normalize and write (B,T,H*DH) bf16
#pragma unroll
  for (int nf = 0; nf < 4; ++nf)
#pragma unroll
    for (int r = 0; r < 8; ++r) {
      const int row = qt * 64 + wave * 16 + mr + r;
      const int d   = nf * 16 + hl;
      att_out[((size_t)b * T_ + row) * (H_ * DH_) + h * DH_ + d] =
          (__bf16)(o[nf][r] / lsum[r]);
    }
}

// ---------------------------------------------------------------------------
// h = LN(concat(conv_out, att_out) + x) * seq_mask  -> f32 + bf16 copies
// ---------------------------------------------------------------------------
__global__ __launch_bounds__(256) void ln1_kernel(
    const float* __restrict__ x, const float* __restrict__ convo,
    const float* __restrict__ atto, const float* __restrict__ g,
    const float* __restrict__ beta, const float* __restrict__ seq_mask,
    float* __restrict__ hf, __bf16* __restrict__ hb) {
  __shared__ float red[256];
  const int row = blockIdx.x, tid = threadIdx.x;
  float v[5];
#pragma unroll
  for (int i = 0; i < 5; ++i) {
    const int c = tid + i * 256;
    float s = x[(size_t)row * EMBED_ + c];
    s += (c < CONV_DIM_) ? convo[(size_t)row * CONV_DIM_ + c]
                         : atto[(size_t)row * D_MODEL_ + (c - CONV_DIM_)];
    v[i] = s;
  }
  red[tid] = v[0] + v[1] + v[2] + v[3] + v[4];
  __syncthreads();
  for (int st = 128; st > 0; st >>= 1) {
    if (tid < st) red[tid] += red[tid + st];
    __syncthreads();
  }
  const float mu = red[0] * (1.0f / EMBED_);
  __syncthreads();
  float var = 0.f;
#pragma unroll
  for (int i = 0; i < 5; ++i) { const float d = v[i] - mu; var += d * d; }
  red[tid] = var;
  __syncthreads();
  for (int st = 128; st > 0; st >>= 1) {
    if (tid < st) red[tid] += red[tid + st];
    __syncthreads();
  }
  const float inv = rsqrtf(red[0] * (1.0f / EMBED_) + EPS_);
  const float sm  = seq_mask[row];
#pragma unroll
  for (int i = 0; i < 5; ++i) {
    const int   c = tid + i * 256;
    const float y = ((v[i] - mu) * inv * g[c] + beta[c]) * sm;
    hf[(size_t)row * EMBED_ + c] = y;
    hb[(size_t)row * EMBED_ + c] = (__bf16)y;
  }
}

// out = LN(ff2 + h) * seq_mask  (f32)
__global__ __launch_bounds__(256) void ln2_kernel(
    const float* __restrict__ ff2, const float* __restrict__ hf,
    const float* __restrict__ g, const float* __restrict__ beta,
    const float* __restrict__ seq_mask, float* __restrict__ out) {
  __shared__ float red[256];
  const int row = blockIdx.x, tid = threadIdx.x;
  float v[5];
#pragma unroll
  for (int i = 0; i < 5; ++i) {
    const int c = tid + i * 256;
    v[i] = ff2[(size_t)row * EMBED_ + c] + hf[(size_t)row * EMBED_ + c];
  }
  red[tid] = v[0] + v[1] + v[2] + v[3] + v[4];
  __syncthreads();
  for (int st = 128; st > 0; st >>= 1) {
    if (tid < st) red[tid] += red[tid + st];
    __syncthreads();
  }
  const float mu = red[0] * (1.0f / EMBED_);
  __syncthreads();
  float var = 0.f;
#pragma unroll
  for (int i = 0; i < 5; ++i) { const float d = v[i] - mu; var += d * d; }
  red[tid] = var;
  __syncthreads();
  for (int st = 128; st > 0; st >>= 1) {
    if (tid < st) red[tid] += red[tid + st];
    __syncthreads();
  }
  const float inv = rsqrtf(red[0] * (1.0f / EMBED_) + EPS_);
  const float sm  = seq_mask[row];
#pragma unroll
  for (int i = 0; i < 5; ++i) {
    const int c = tid + i * 256;
    out[(size_t)row * EMBED_ + c] = ((v[i] - mu) * inv * g[c] + beta[c]) * sm;
  }
}

// ---------------------------------------------------------------------------
extern "C" void kernel_launch(void* const* d_in, const int* in_sizes, int n_in,
                              void* d_out, int out_size, void* d_ws,
                              size_t ws_size, hipStream_t stream) {
  (void)in_sizes; (void)n_in; (void)out_size; (void)ws_size;
  const float* x        = (const float*)d_in[0];
  const int*   att_mask = (const int*)d_in[1];
  const float* seq_mask = (const float*)d_in[2];
  const float* conv_w   = (const float*)d_in[3];
  const float* wq = (const float*)d_in[4];  const float* bq = (const float*)d_in[5];
  const float* wk = (const float*)d_in[6];  const float* bk = (const float*)d_in[7];
  const float* wv = (const float*)d_in[8];  const float* bv = (const float*)d_in[9];
  const float* wo = (const float*)d_in[10]; const float* bo = (const float*)d_in[11];
  const float* w1 = (const float*)d_in[12]; const float* b1 = (const float*)d_in[13];
  const float* w2 = (const float*)d_in[14]; const float* b2 = (const float*)d_in[15];
  const float* g1 = (const float*)d_in[16]; const float* be1 = (const float*)d_in[17];
  const float* g2 = (const float*)d_in[18]; const float* be2 = (const float*)d_in[19];

  char*  ws  = (char*)d_ws;
  size_t off = 0;
  auto bump = [&](size_t bytes) -> void* {
    void* p = ws + off;
    off += (bytes + 255) & ~(size_t)255;
    return p;
  };

  const size_t MT = (size_t)B_ * T_;  // 8192 rows
  __bf16* att_in_bf = (__bf16*)bump(MT * D_MODEL_ * 2);
  __bf16* wqt = (__bf16*)bump((size_t)D_MODEL_ * D_MODEL_ * 2);
  __bf16* wkt = (__bf16*)bump((size_t)D_MODEL_ * D_MODEL_ * 2);
  __bf16* wvt = (__bf16*)bump((size_t)D_MODEL_ * D_MODEL_ * 2);
  __bf16* wot = (__bf16*)bump((size_t)D_MODEL_ * D_MODEL_ * 2);
  __bf16* w1t = (__bf16*)bump((size_t)EMBED_ * D_FF_ * 2);
  __bf16* w2t = (__bf16*)bump((size_t)D_FF_ * EMBED_ * 2);
  __bf16* qb  = (__bf16*)bump(MT * D_MODEL_ * 2);
  __bf16* kb  = (__bf16*)bump(MT * D_MODEL_ * 2);
  __bf16* vb  = (__bf16*)bump(MT * D_MODEL_ * 2);
  __bf16* attb = (__bf16*)bump(MT * D_MODEL_ * 2);
  float*  convo = (float*)bump(MT * CONV_DIM_ * 4);
  float*  atto  = (float*)bump(MT * D_MODEL_ * 4);
  float*  hf    = (float*)bump(MT * EMBED_ * 4);
  __bf16* hb    = (__bf16*)bump(MT * EMBED_ * 2);
  __bf16* ffb   = (__bf16*)bump(MT * D_FF_ * 2);
  float*  ff2   = (float*)bump(MT * EMBED_ * 4);

  // weight precondition (bf16, transposed to N x K)
  transpose_convert_kernel<<<dim3(D_MODEL_ / 32, D_MODEL_ / 32), 256, 0, stream>>>(wq, wqt, D_MODEL_, D_MODEL_);
  transpose_convert_kernel<<<dim3(D_MODEL_ / 32, D_MODEL_ / 32), 256, 0, stream>>>(wk, wkt, D_MODEL_, D_MODEL_);
  transpose_convert_kernel<<<dim3(D_MODEL_ / 32, D_MODEL_ / 32), 256, 0, stream>>>(wv, wvt, D_MODEL_, D_MODEL_);
  transpose_convert_kernel<<<dim3(D_MODEL_ / 32, D_MODEL_ / 32), 256, 0, stream>>>(wo, wot, D_MODEL_, D_MODEL_);
  transpose_convert_kernel<<<dim3(D_FF_ / 32, EMBED_ / 32), 256, 0, stream>>>(w1, w1t, EMBED_, D_FF_);
  transpose_convert_kernel<<<dim3(EMBED_ / 32, D_FF_ / 32), 256, 0, stream>>>(w2, w2t, D_FF_, EMBED_);
  convert_attin_kernel<<<(unsigned)(MT * D_MODEL_ / 256), 256, 0, stream>>>(x, att_in_bf);
  conv_kernel<<<(unsigned)(MT * CONV_DIM_ / 256), 256, 0, stream>>>(x, conv_w, convo);

  // QKV projections -> (B,H,T,DH) bf16
  const dim3 gqkv(D_MODEL_ / 128, (unsigned)(MT / 128));
  gemm_bf16_kernel<<<gqkv, 256, 0, stream>>>(att_in_bf, wqt, bq, qb, (int)MT, D_MODEL_, D_MODEL_, 0);
  gemm_bf16_kernel<<<gqkv, 256, 0, stream>>>(att_in_bf, wkt, bk, kb, (int)MT, D_MODEL_, D_MODEL_, 0);
  gemm_bf16_kernel<<<gqkv, 256, 0, stream>>>(att_in_bf, wvt, bv, vb, (int)MT, D_MODEL_, D_MODEL_, 0);

  attention_kernel<<<dim3(T_ / 64, H_, B_), 128, 0, stream>>>(qb, kb, vb, att_mask, attb);

  // output projection -> f32
  gemm_bf16_kernel<<<gqkv, 256, 0, stream>>>(attb, wot, bo, atto, (int)MT, D_MODEL_, D_MODEL_, 1);

  ln1_kernel<<<(unsigned)MT, 256, 0, stream>>>(x, convo, atto, g1, be1, seq_mask, hf, hb);

  // FFN
  gemm_bf16_kernel<<<dim3(D_FF_ / 128, (unsigned)(MT / 128)), 256, 0, stream>>>(hb, w1t, b1, ffb, (int)MT, D_FF_, EMBED_, 2);
  gemm_bf16_kernel<<<dim3(EMBED_ / 128, (unsigned)(MT / 128)), 256, 0, stream>>>(ffb, w2t, b2, ff2, (int)MT, EMBED_, D_FF_, 1);

  ln2_kernel<<<(unsigned)MT, 256, 0, stream>>>(ff2, hf, g2, be2, seq_mask, (float*)d_out);
}
